// RSSM_47648367182020
// MI455X (gfx1250) — matrix operand order repaired and checked by reading, not compiled, not run on hardware
//
#include <hip/hip_runtime.h>

// ---------------------------------------------------------------------------
// RSSM forward for MI455X (gfx1250): bf16 WMMA 16x16x32, f32 accumulate.
// LDS-staged weight tiles (shared by all 8 waves of a block -> 8x less L2
// weight traffic), 2x2 register blocking, fused GRU + fused post/rsample.
// ---------------------------------------------------------------------------

typedef __bf16 bf16_t;
typedef __attribute__((ext_vector_type(16))) __bf16 v16bf;
typedef __attribute__((ext_vector_type(8)))  __bf16 v8bf;
typedef __attribute__((ext_vector_type(4)))  __bf16 v4bf;
typedef __attribute__((ext_vector_type(8)))  float  v8f;

#define T_STEPS 64
#define B_SZ    256
#define E_SZ    1536
#define A_SZ    128
#define D_SZ    2048
#define Z_SZ    512
#define H_SZ    1024

#define KC    256            // K-chunk staged per barrier round
#define LDSK  264            // padded LDS row stride (528B) -> conflict-free

__device__ __forceinline__ v8f wmma_bf16(v16bf a, v16bf b, v8f c) {
  return __builtin_amdgcn_wmma_f32_16x16x32_bf16(
      false, a, false, b, (short)0, c, false, false);
}

// A-fragment: 16x32 bf16 tile of row-major [.,K] matrix (ISA 7.12.2 layout).
__device__ __forceinline__ v16bf load_a_frag(const bf16_t* X, int ld, int k0, int lane) {
  const int m  = lane & 15;
  const int kb = (lane >> 4) << 3;
  const bf16_t* p = X + (size_t)m * ld + k0 + kb;
  v8bf lo = *(const v8bf*)p;         // K = kb .. kb+7
  v8bf hi = *(const v8bf*)(p + 16);  // K = 16+kb .. 16+kb+7
  v16bf a;
#pragma unroll
  for (int i = 0; i < 8; ++i) { a[i] = lo[i]; a[i + 8] = hi[i]; }
  return a;
}

// B-fragment from an LDS-staged weight tile (rows = output cols, LDSK stride).
__device__ __forceinline__ v16bf load_b_frag_lds(const bf16_t* s, int k0, int lane) {
  const int n  = lane & 15;
  const int kb = (lane >> 4) << 3;
  const bf16_t* p = s + n * LDSK + k0 + kb;
  v8bf lo = *(const v8bf*)p;
  v8bf hi = *(const v8bf*)(p + 16);
  v16bf b;
#pragma unroll
  for (int i = 0; i < 8; ++i) { b[i] = lo[i]; b[i + 8] = hi[i]; }
  return b;
}

// Cooperative stage of a [rows x kcur] bf16 weight tile into LDS (256 thr).
__device__ __forceinline__ void stage_w(const bf16_t* __restrict__ g, int ldg,
                                        bf16_t* s, int rows, int kcur, int tid) {
  const int kv  = kcur >> 3;           // b128 vectors per row
  const int vec = rows * kv;
  for (int i = tid; i < vec; i += 256) {
    const int r = i / kv;
    const int c = (i - r * kv) << 3;
    *(v8bf*)(s + r * LDSK + c) = *(const v8bf*)(g + (size_t)r * ldg + c);
  }
}

__device__ __forceinline__ float elu1(float x) { return x > 0.f ? x : __expf(x) - 1.f; }
__device__ __forceinline__ float sigm(float x) { return 1.f / (1.f + __expf(-x)); }

// ---------------------------------------------------------------------------
// f32 -> bf16 conversion (4-wide, grid-stride)
// ---------------------------------------------------------------------------
__global__ void k_cvt4(const float4* __restrict__ src, v4bf* __restrict__ dst, int n4) {
  for (int i = blockIdx.x * blockDim.x + threadIdx.x; i < n4;
       i += gridDim.x * blockDim.x) {
    float4 v = src[i];
    v4bf o = { (__bf16)v.x, (__bf16)v.y, (__bf16)v.z, (__bf16)v.w };
    dst[i] = o;
  }
}

// ---------------------------------------------------------------------------
// LDS-staged 2x2-blocked GEMM: out[B,N] = act(A[B,K] @ W^T + bias)
// Block: 256 rows x 32 cols; wave w owns rows 32w..32w+31 (4 accums).
// Weight tile read from L2 once per block, shared via LDS.
// grid = N/32 blocks x 256 threads.
// ---------------------------------------------------------------------------
template <bool ELU, bool BF16OUT>
__global__ void k_gemm22(const bf16_t* __restrict__ A,
                         const bf16_t* __restrict__ W,
                         const float*  __restrict__ bias,
                         void* __restrict__ outp, int K, int N) {
  __shared__ bf16_t sw[32 * LDSK];
  const int tid  = threadIdx.x;
  const int lane = tid & 31;
  const int tm   = (tid >> 5) << 5;     // wave's 32-row group
  const int tn   = blockIdx.x << 5;     // block's 32-col group
  const bf16_t* a0 = A + (size_t)tm * K;
  const bf16_t* a1 = a0 + (size_t)16 * K;
  v8f c00 = {}, c01 = {}, c10 = {}, c11 = {};
  for (int kc = 0; kc < K; kc += KC) {
    const int kcur = (K - kc < KC) ? (K - kc) : KC;
    __syncthreads();
    stage_w(W + (size_t)tn * K + kc, K, sw, 32, kcur, tid);
    __syncthreads();
    for (int k = 0; k < kcur; k += 32) {
      v16bf fa0 = load_a_frag(a0, K, kc + k, lane);
      v16bf fa1 = load_a_frag(a1, K, kc + k, lane);
      v16bf fb0 = load_b_frag_lds(sw, k, lane);
      v16bf fb1 = load_b_frag_lds(sw + 16 * LDSK, k, lane);
      c00 = wmma_bf16(fa0, fb0, c00);
      c01 = wmma_bf16(fa0, fb1, c01);
      c10 = wmma_bf16(fa1, fb0, c10);
      c11 = wmma_bf16(fa1, fb1, c11);
    }
  }
  const int cn0 = tn + (lane & 15), cn1 = cn0 + 16;
  const float bv0 = bias[cn0], bv1 = bias[cn1];
  const int r0 = tm + ((lane >> 4) << 3);
#pragma unroll
  for (int v = 0; v < 8; ++v) {
    float x00 = c00[v] + bv0, x01 = c01[v] + bv1;
    float x10 = c10[v] + bv0, x11 = c11[v] + bv1;
    if (ELU) { x00 = elu1(x00); x01 = elu1(x01); x10 = elu1(x10); x11 = elu1(x11); }
    const size_t i0 = (size_t)(r0 + v) * N, i1 = (size_t)(r0 + 16 + v) * N;
    if (BF16OUT) {
      bf16_t* o = (bf16_t*)outp;
      o[i0 + cn0] = (__bf16)x00; o[i0 + cn1] = (__bf16)x01;
      o[i1 + cn0] = (__bf16)x10; o[i1 + cn1] = (__bf16)x11;
    } else {
      float* o = (float*)outp;
      o[i0 + cn0] = x00; o[i0 + cn1] = x01;
      o[i1 + cn0] = x10; o[i1 + cn1] = x11;
    }
  }
}

// ---------------------------------------------------------------------------
// Dual-input LDS-staged 2x2 GEMM + ELU -> bf16 (za / he combinations):
// out[B,N] = elu(A1[B,K1] @ W1^T + A2[B,K2] @ W2^T + bias)
// ---------------------------------------------------------------------------
__global__ void k_dual22(const bf16_t* __restrict__ A1, int K1,
                         const bf16_t* __restrict__ A2, int K2,
                         const bf16_t* __restrict__ W1,
                         const bf16_t* __restrict__ W2,
                         const float*  __restrict__ bias,
                         bf16_t* __restrict__ outbf, int N) {
  __shared__ bf16_t sw[32 * LDSK];
  const int tid  = threadIdx.x;
  const int lane = tid & 31;
  const int tm   = (tid >> 5) << 5;
  const int tn   = blockIdx.x << 5;
  v8f c00 = {}, c01 = {}, c10 = {}, c11 = {};
#pragma unroll
  for (int phase = 0; phase < 2; ++phase) {
    const bf16_t* A = phase ? A2 : A1;
    const bf16_t* W = phase ? W2 : W1;
    const int     K = phase ? K2 : K1;
    const bf16_t* a0 = A + (size_t)tm * K;
    const bf16_t* a1 = a0 + (size_t)16 * K;
    for (int kc = 0; kc < K; kc += KC) {
      const int kcur = (K - kc < KC) ? (K - kc) : KC;
      __syncthreads();
      stage_w(W + (size_t)tn * K + kc, K, sw, 32, kcur, tid);
      __syncthreads();
      for (int k = 0; k < kcur; k += 32) {
        v16bf fa0 = load_a_frag(a0, K, kc + k, lane);
        v16bf fa1 = load_a_frag(a1, K, kc + k, lane);
        v16bf fb0 = load_b_frag_lds(sw, k, lane);
        v16bf fb1 = load_b_frag_lds(sw + 16 * LDSK, k, lane);
        c00 = wmma_bf16(fa0, fb0, c00);
        c01 = wmma_bf16(fa0, fb1, c01);
        c10 = wmma_bf16(fa1, fb0, c10);
        c11 = wmma_bf16(fa1, fb1, c11);
      }
    }
  }
  const int cn0 = tn + (lane & 15), cn1 = cn0 + 16;
  const float bv0 = bias[cn0], bv1 = bias[cn1];
  const int r0 = tm + ((lane >> 4) << 3);
#pragma unroll
  for (int v = 0; v < 8; ++v) {
    const size_t i0 = (size_t)(r0 + v) * N, i1 = (size_t)(r0 + 16 + v) * N;
    outbf[i0 + cn0] = (__bf16)elu1(c00[v] + bv0);
    outbf[i0 + cn1] = (__bf16)elu1(c01[v] + bv1);
    outbf[i1 + cn0] = (__bf16)elu1(c10[v] + bv0);
    outbf[i1 + cn1] = (__bf16)elu1(c11[v] + bv1);
  }
}

// ---------------------------------------------------------------------------
// Fused GRU step, LDS-staged: block = 256 rows x 16 cols of h_new; per chunk
// the 3 gate weight tiles (r/z/n) are staged once and shared by 8 waves.
// Each wave: 2 row tiles, 12 accumulators; per k-step 2 A-frags + 3 LDS
// B-frags -> 6 WMMA. grid = D/16 = 128 blocks.
// ---------------------------------------------------------------------------
__global__ void k_gru(const bf16_t* __restrict__ za_bf,
                      const bf16_t* __restrict__ h_bf_in,
                      const float*  __restrict__ h_f32_in,
                      const bf16_t* __restrict__ w_ih,
                      const bf16_t* __restrict__ w_hh,
                      const float*  __restrict__ b_ih,
                      const float*  __restrict__ b_hh,
                      float*  __restrict__ h_f32_out,
                      bf16_t* __restrict__ h_bf_out,
                      float*  __restrict__ states_h_t) {
  __shared__ bf16_t sw[3 * 16 * LDSK];
  const int tid  = threadIdx.x;
  const int lane = tid & 31;
  const int tm   = (tid >> 5) << 5;     // wave's 32-row group
  const int tn   = blockIdx.x << 4;     // block's 16-col tile of D

  v8f ir[2] = {{}, {}}, iz[2] = {{}, {}}, in[2] = {{}, {}};
  v8f hr[2] = {{}, {}}, hz[2] = {{}, {}}, hn[2] = {{}, {}};

  // phase 1: gi = za @ w_ih^T  (K = H)
  {
    const bf16_t* a_base = za_bf + (size_t)tm * H_SZ;
    const bf16_t* wr = w_ih + (size_t)tn * H_SZ;
    for (int kc = 0; kc < H_SZ; kc += KC) {
      __syncthreads();
      stage_w(wr + kc,                               H_SZ, sw,              16, KC, tid);
      stage_w(wr + (size_t)D_SZ * H_SZ + kc,         H_SZ, sw + 16 * LDSK,  16, KC, tid);
      stage_w(wr + (size_t)2 * D_SZ * H_SZ + kc,     H_SZ, sw + 32 * LDSK,  16, KC, tid);
      __syncthreads();
      for (int k = 0; k < KC; k += 32) {
        v16bf a0 = load_a_frag(a_base, H_SZ, kc + k, lane);
        v16bf a1 = load_a_frag(a_base + (size_t)16 * H_SZ, H_SZ, kc + k, lane);
        v16bf br = load_b_frag_lds(sw,             k, lane);
        v16bf bz = load_b_frag_lds(sw + 16 * LDSK, k, lane);
        v16bf bn = load_b_frag_lds(sw + 32 * LDSK, k, lane);
        ir[0] = wmma_bf16(a0, br, ir[0]);  ir[1] = wmma_bf16(a1, br, ir[1]);
        iz[0] = wmma_bf16(a0, bz, iz[0]);  iz[1] = wmma_bf16(a1, bz, iz[1]);
        in[0] = wmma_bf16(a0, bn, in[0]);  in[1] = wmma_bf16(a1, bn, in[1]);
      }
    }
  }
  // phase 2: gh = h_old @ w_hh^T  (K = D)
  {
    const bf16_t* a_base = h_bf_in + (size_t)tm * D_SZ;
    const bf16_t* ur = w_hh + (size_t)tn * D_SZ;
    for (int kc = 0; kc < D_SZ; kc += KC) {
      __syncthreads();
      stage_w(ur + kc,                               D_SZ, sw,              16, KC, tid);
      stage_w(ur + (size_t)D_SZ * D_SZ + kc,         D_SZ, sw + 16 * LDSK,  16, KC, tid);
      stage_w(ur + (size_t)2 * D_SZ * D_SZ + kc,     D_SZ, sw + 32 * LDSK,  16, KC, tid);
      __syncthreads();
      for (int k = 0; k < KC; k += 32) {
        v16bf a0 = load_a_frag(a_base, D_SZ, kc + k, lane);
        v16bf a1 = load_a_frag(a_base + (size_t)16 * D_SZ, D_SZ, kc + k, lane);
        v16bf br = load_b_frag_lds(sw,             k, lane);
        v16bf bz = load_b_frag_lds(sw + 16 * LDSK, k, lane);
        v16bf bn = load_b_frag_lds(sw + 32 * LDSK, k, lane);
        hr[0] = wmma_bf16(a0, br, hr[0]);  hr[1] = wmma_bf16(a1, br, hr[1]);
        hz[0] = wmma_bf16(a0, bz, hz[0]);  hz[1] = wmma_bf16(a1, bz, hz[1]);
        hn[0] = wmma_bf16(a0, bn, hn[0]);  hn[1] = wmma_bf16(a1, bn, hn[1]);
      }
    }
  }

  const int col = tn + (lane & 15);
  const float bir = b_ih[col], biz = b_ih[D_SZ + col], bin = b_ih[2 * D_SZ + col];
  const float bhr = b_hh[col], bhz = b_hh[D_SZ + col], bhn = b_hh[2 * D_SZ + col];
#pragma unroll
  for (int g = 0; g < 2; ++g) {
    const int rbase = tm + (g << 4) + ((lane >> 4) << 3);
#pragma unroll
    for (int v = 0; v < 8; ++v) {
      const size_t idx = (size_t)(rbase + v) * D_SZ + col;
      float r    = sigm(ir[g][v] + bir + hr[g][v] + bhr);
      float zg   = sigm(iz[g][v] + biz + hz[g][v] + bhz);
      float nn   = tanhf(in[g][v] + bin + r * (hn[g][v] + bhn));
      float hold = h_f32_in[idx];
      float hnew = (1.f - zg) * nn + zg * hold;
      h_f32_out[idx]  = hnew;
      h_bf_out[idx]   = (__bf16)hnew;
      states_h_t[idx] = hnew;
    }
  }
}

// ---------------------------------------------------------------------------
// Fused posterior head + rsample, LDS-staged: block = 256 rows x column pair
// (n..n+15 mean, Z+n..Z+n+15 std). Wave: 2 row tiles x (mean,std) = 4 accums.
//   posts[t] = he @ post_w^T + b ; std = 2*sigmoid(raw)+0.1
//   samples[t] = mean + std*eps  ; z_bf = bf16(sample)
// grid = Z/16 = 32 blocks.
// ---------------------------------------------------------------------------
__global__ void k_post_sample(const bf16_t* __restrict__ he_bf,
                              const bf16_t* __restrict__ w_post,
                              const float*  __restrict__ b_post,
                              const float*  __restrict__ eps_t,
                              float* __restrict__ posts_t,
                              float* __restrict__ samples_t,
                              bf16_t* __restrict__ z_bf) {
  __shared__ bf16_t sw[2 * 16 * LDSK];
  const int tid  = threadIdx.x;
  const int lane = tid & 31;
  const int tm   = (tid >> 5) << 5;
  const int tn   = blockIdx.x << 4;     // mean col tile; std tile at Z+tn
  const bf16_t* a_base = he_bf + (size_t)tm * H_SZ;
  v8f cm[2] = {{}, {}}, cs[2] = {{}, {}};
  for (int kc = 0; kc < H_SZ; kc += KC) {
    __syncthreads();
    stage_w(w_post + (size_t)tn * H_SZ + kc,          H_SZ, sw,             16, KC, tid);
    stage_w(w_post + (size_t)(Z_SZ + tn) * H_SZ + kc, H_SZ, sw + 16 * LDSK, 16, KC, tid);
    __syncthreads();
    for (int k = 0; k < KC; k += 32) {
      v16bf a0 = load_a_frag(a_base, H_SZ, kc + k, lane);
      v16bf a1 = load_a_frag(a_base + (size_t)16 * H_SZ, H_SZ, kc + k, lane);
      v16bf bm = load_b_frag_lds(sw,             k, lane);
      v16bf bs = load_b_frag_lds(sw + 16 * LDSK, k, lane);
      cm[0] = wmma_bf16(a0, bm, cm[0]);  cm[1] = wmma_bf16(a1, bm, cm[1]);
      cs[0] = wmma_bf16(a0, bs, cs[0]);  cs[1] = wmma_bf16(a1, bs, cs[1]);
    }
  }
  const int col = tn + (lane & 15);
  const float bm = b_post[col], bs = b_post[Z_SZ + col];
#pragma unroll
  for (int g = 0; g < 2; ++g) {
    const int rbase = tm + (g << 4) + ((lane >> 4) << 3);
#pragma unroll
    for (int v = 0; v < 8; ++v) {
      const int row = rbase + v;
      const float mean = cm[g][v] + bm;
      const float sraw = cs[g][v] + bs;
      posts_t[(size_t)row * (2 * Z_SZ) + col]        = mean;
      posts_t[(size_t)row * (2 * Z_SZ) + Z_SZ + col] = sraw;
      const float std = 2.0f * sigm(sraw) + 0.1f;
      const float s   = mean + std * eps_t[(size_t)row * Z_SZ + col];
      samples_t[(size_t)row * Z_SZ + col] = s;
      z_bf[(size_t)row * Z_SZ + col]      = (__bf16)s;
    }
  }
}

// ---------------------------------------------------------------------------
// Host-side launch
// ---------------------------------------------------------------------------
extern "C" void kernel_launch(void* const* d_in, const int* in_sizes, int n_in,
                              void* d_out, int out_size, void* d_ws, size_t ws_size,
                              hipStream_t stream) {
  const float* embedded    = (const float*)d_in[0];   // [T,B,E]
  const float* action      = (const float*)d_in[1];   // [T,B,A]
  const float* z_in        = (const float*)d_in[2];   // [B,Z]
  const float* h_in        = (const float*)d_in[3];   // [B,D]
  const float* eps         = (const float*)d_in[4];   // [T,B,Z]
  const float* za_w1       = (const float*)d_in[5];
  const float* za_b1       = (const float*)d_in[6];
  const float* za_w2       = (const float*)d_in[7];
  const float* he_w1       = (const float*)d_in[8];
  const float* he_b1       = (const float*)d_in[9];
  const float* he_w2       = (const float*)d_in[10];
  const float* prior_h_w   = (const float*)d_in[11];
  const float* prior_h_b   = (const float*)d_in[12];
  const float* prior_out_w = (const float*)d_in[13];
  const float* prior_out_b = (const float*)d_in[14];
  const float* post_w      = (const float*)d_in[15];
  const float* post_b      = (const float*)d_in[16];
  const float* gru_w_ih    = (const float*)d_in[17];
  const float* gru_w_hh    = (const float*)d_in[18];
  const float* gru_b_ih    = (const float*)d_in[19];
  const float* gru_b_hh    = (const float*)d_in[20];

  float* out = (float*)d_out;
  const size_t P_POSTS   = (size_t)T_STEPS * B_SZ * 2 * Z_SZ;
  const size_t P_SAMPLES = 2 * P_POSTS;
  const size_t P_STATES  = P_SAMPLES + (size_t)T_STEPS * B_SZ * Z_SZ;
  float* priors_base  = out;
  float* posts_base   = out + P_POSTS;
  float* samples_base = out + P_SAMPLES;
  float* states_base  = out + P_STATES;

  // ---- workspace layout (256B aligned regions) ----
  char* ws = (char*)d_ws;
  size_t off = 0;
  auto abf = [&](size_t n) -> bf16_t* {
    bf16_t* p = (bf16_t*)(ws + off);
    off = (off + n * sizeof(bf16_t) + 255) & ~(size_t)255;
    return p;
  };
  auto af32 = [&](size_t n) -> float* {
    float* p = (float*)(ws + off);
    off = (off + n * sizeof(float) + 255) & ~(size_t)255;
    return p;
  };
  bf16_t* W_ZA1 = abf((size_t)H_SZ * Z_SZ);
  bf16_t* W_ZA2 = abf((size_t)H_SZ * A_SZ);
  bf16_t* W_HE1 = abf((size_t)H_SZ * D_SZ);
  bf16_t* W_HE2 = abf((size_t)H_SZ * E_SZ);
  bf16_t* W_PH  = abf((size_t)H_SZ * D_SZ);
  bf16_t* W_PO  = abf((size_t)2 * Z_SZ * H_SZ);
  bf16_t* W_PS  = abf((size_t)2 * Z_SZ * H_SZ);
  bf16_t* W_IH  = abf((size_t)3 * D_SZ * H_SZ);
  bf16_t* W_HH  = abf((size_t)3 * D_SZ * D_SZ);
  bf16_t* a_bf  = abf((size_t)T_STEPS * B_SZ * A_SZ);   // full action, bf16
  bf16_t* e_bf  = abf((size_t)B_SZ * E_SZ);             // per-step embedded slice
  bf16_t* z_bf  = abf((size_t)B_SZ * Z_SZ);
  bf16_t* za_bf = abf((size_t)B_SZ * H_SZ);
  bf16_t* he_bf = abf((size_t)B_SZ * H_SZ);
  bf16_t* ph_bf = abf((size_t)B_SZ * H_SZ);
  bf16_t* h_bf0 = abf((size_t)B_SZ * D_SZ);
  bf16_t* h_bf1 = abf((size_t)B_SZ * D_SZ);
  float*  h_f0  = af32((size_t)B_SZ * D_SZ);
  float*  h_f1  = af32((size_t)B_SZ * D_SZ);
  bf16_t* h_bf[2] = { h_bf0, h_bf1 };
  float*  h_f[2]  = { h_f0, h_f1 };
  (void)ws_size; (void)in_sizes; (void)n_in; (void)out_size;

  auto cvt = [&](const float* s, bf16_t* d, size_t n) {
    int n4 = (int)(n / 4);
    int blocks = (n4 + 255) / 256;
    if (blocks > 4096) blocks = 4096;
    k_cvt4<<<blocks, 256, 0, stream>>>((const float4*)s, (v4bf*)d, n4);
  };

  // ---- per-call preamble: f32 -> bf16 ----
  cvt(za_w1, W_ZA1, (size_t)H_SZ * Z_SZ);
  cvt(za_w2, W_ZA2, (size_t)H_SZ * A_SZ);
  cvt(he_w1, W_HE1, (size_t)H_SZ * D_SZ);
  cvt(he_w2, W_HE2, (size_t)H_SZ * E_SZ);
  cvt(prior_h_w, W_PH, (size_t)H_SZ * D_SZ);
  cvt(prior_out_w, W_PO, (size_t)2 * Z_SZ * H_SZ);
  cvt(post_w, W_PS, (size_t)2 * Z_SZ * H_SZ);
  cvt(gru_w_ih, W_IH, (size_t)3 * D_SZ * H_SZ);
  cvt(gru_w_hh, W_HH, (size_t)3 * D_SZ * D_SZ);
  cvt(action, a_bf, (size_t)T_STEPS * B_SZ * A_SZ);
  cvt(z_in, z_bf, (size_t)B_SZ * Z_SZ);
  cvt(h_in, h_bf[0], (size_t)B_SZ * D_SZ);
  hipMemcpyAsync(h_f[0], h_in, (size_t)B_SZ * D_SZ * sizeof(float),
                 hipMemcpyDeviceToDevice, stream);

  // ---- scan over T ----
  for (int t = 0; t < T_STEPS; ++t) {
    const int hrd = t & 1;
    const int hwr = 1 - hrd;
    const float*  emb_t  = embedded + (size_t)t * B_SZ * E_SZ;
    const bf16_t* act_bt = a_bf + (size_t)t * B_SZ * A_SZ;
    const float*  eps_t  = eps + (size_t)t * B_SZ * Z_SZ;
    float* priors_t  = priors_base  + (size_t)t * B_SZ * 2 * Z_SZ;
    float* posts_t   = posts_base   + (size_t)t * B_SZ * 2 * Z_SZ;
    float* samples_t = samples_base + (size_t)t * B_SZ * Z_SZ;
    float* states_t  = states_base  + (size_t)t * B_SZ * D_SZ;

    // embedded slice -> bf16 (consumed by k_dual22 'he' below)
    cvt(emb_t, e_bf, (size_t)B_SZ * E_SZ);

    // za = elu(z @ za_w1^T + a @ za_w2^T + b1)              [B,H]
    k_dual22<<<H_SZ / 32, 256, 0, stream>>>(
        z_bf, Z_SZ, act_bt, A_SZ, W_ZA1, W_ZA2, za_b1, za_bf, H_SZ);

    // h_new = GRU(za, h_old)                                [B,D]
    k_gru<<<D_SZ / 16, 256, 0, stream>>>(
        za_bf, h_bf[hrd], h_f[hrd], W_IH, W_HH, gru_b_ih, gru_b_hh,
        h_f[hwr], h_bf[hwr], states_t);

    // he = elu(h_new @ he_w1^T + e @ he_w2^T + b1)          [B,H]
    k_dual22<<<H_SZ / 32, 256, 0, stream>>>(
        h_bf[hwr], D_SZ, e_bf, E_SZ, W_HE1, W_HE2, he_b1, he_bf, H_SZ);

    // post head + rsample (fused)                           [B,2Z],[B,Z]
    k_post_sample<<<Z_SZ / 16, 256, 0, stream>>>(
        he_bf, W_PS, post_b, eps_t, posts_t, samples_t, z_bf);

    // prior = elu(h_new @ prior_h_w^T + b) @ prior_out_w^T + b
    k_gemm22<true, true><<<H_SZ / 32, 256, 0, stream>>>(
        h_bf[hwr], W_PH, prior_h_b, (void*)ph_bf, D_SZ, H_SZ);
    k_gemm22<false, false><<<(2 * Z_SZ) / 32, 256, 0, stream>>>(
        ph_bf, W_PO, prior_out_b, (void*)priors_t, H_SZ, 2 * Z_SZ);
  }
}
